// LSTMDecoder_22093311770648
// MI455X (gfx1250) — matrix operand order repaired and checked by reading, not compiled, not run on hardware
//
#include <hip/hip_runtime.h>
#include <cstdint>
#include <cstddef>

// ---------------------------------------------------------------------------
// Types for CDNA5 WMMA (wave32): v_wmma_f32_16x16x32_bf16
// ---------------------------------------------------------------------------
typedef __bf16 bf16_t;
typedef __attribute__((ext_vector_type(16))) __bf16 v16bf;
typedef __attribute__((ext_vector_type(8)))  __bf16 v8bf;
typedef __attribute__((ext_vector_type(8)))  float  v8f;

union V16U { v16bf v; v8bf h[2]; };

__device__ __forceinline__ unsigned short f2bf(float f) {
  unsigned int u = __builtin_bit_cast(unsigned int, f);
  unsigned int r = u + 0x7FFFu + ((u >> 16) & 1u);   // round-to-nearest-even
  return (unsigned short)(r >> 16);
}
__device__ __forceinline__ float bf2f(unsigned short s) {
  unsigned int u = ((unsigned int)s) << 16;
  return __builtin_bit_cast(float, u);
}
__device__ __forceinline__ float sigmoidf_(float x) {
  return 1.0f / (1.0f + __expf(-x));
}

// ---------------------------------------------------------------------------
// WMMA GEMM:  C(MxN) = A(MxK) * W(NxK)^T + bias   (A, W in bf16; acc fp32)
// One wave computes a 32x32 patch = 2x2 WMMA tiles (4 independent acc chains;
// 2 A-frag + 2 B-frag loads feed 4 WMMAs -> 1KB/wmma instead of 2KB/wmma).
// Grid must cover tile-pairs exactly (EXEC all ones for WMMA).
// MODE 0: store fp32 (gates).  MODE 1: store bf16 (enc_proj).
// MODE 2: tanh epilogue, store fp32 (output slab) + bf16 (next-step staging).
// ---------------------------------------------------------------------------
template<int MODE>
__global__ __launch_bounds__(256)
void wmma_gemm(const unsigned short* __restrict__ A, int lda,
               const unsigned short* __restrict__ W, int ldw,
               const float* __restrict__ bias,
               float* __restrict__ outF, int ldo,
               unsigned short* __restrict__ outB, int ldob,
               int M, int N, int K)
{
  const int wave = (blockIdx.x * blockDim.x + threadIdx.x) >> 5;
  const int lane = threadIdx.x & 31;
  const int mP = M >> 5;                  // 32-row patches
  const int nP = N >> 5;                  // 32-col patches
  const int mp = wave % mP;
  const int np = wave / mP;
  if (np >= nP) return;                   // wave-uniform (lane-independent)

  const int m0 = mp << 5, n0 = np << 5;
  const int lm = lane & 15, lh = lane >> 4;

  // A frag: lane holds row m+lm; elems 0-7 = K(k0+lh*8..), elems 8-15 = +16
  const unsigned short* aRow0 = A + (size_t)(m0 + lm) * lda + lh * 8;
  const unsigned short* aRow1 = aRow0 + (size_t)16 * lda;
  // B frag: lane holds col n+lm; elems 0-7 = K(k0+lh*8..), elems 8-15 = +16
  const unsigned short* wRow0 = W + (size_t)(n0 + lm) * ldw + lh * 8;
  const unsigned short* wRow1 = wRow0 + (size_t)16 * ldw;

  v8f acc00 = {}, acc01 = {}, acc10 = {}, acc11 = {};
  #pragma unroll 2
  for (int k0 = 0; k0 < K; k0 += 32) {
    V16U a0, a1, b0, b1;
    a0.h[0] = *(const v8bf*)(const void*)(aRow0 + k0);
    a0.h[1] = *(const v8bf*)(const void*)(aRow0 + k0 + 16);
    a1.h[0] = *(const v8bf*)(const void*)(aRow1 + k0);
    a1.h[1] = *(const v8bf*)(const void*)(aRow1 + k0 + 16);
    b0.h[0] = *(const v8bf*)(const void*)(wRow0 + k0);
    b0.h[1] = *(const v8bf*)(const void*)(wRow0 + k0 + 16);
    b1.h[0] = *(const v8bf*)(const void*)(wRow1 + k0);
    b1.h[1] = *(const v8bf*)(const void*)(wRow1 + k0 + 16);
    acc00 = __builtin_amdgcn_wmma_f32_16x16x32_bf16(
        false, a0.v, false, b0.v, (short)0, acc00, false, false);
    acc01 = __builtin_amdgcn_wmma_f32_16x16x32_bf16(
        false, a0.v, false, b1.v, (short)0, acc01, false, false);
    acc10 = __builtin_amdgcn_wmma_f32_16x16x32_bf16(
        false, a1.v, false, b0.v, (short)0, acc10, false, false);
    acc11 = __builtin_amdgcn_wmma_f32_16x16x32_bf16(
        false, a1.v, false, b1.v, (short)0, acc11, false, false);
  }

  // Epilogue: C layout -> reg r maps to row (m + r + 8*lh), col (n + lm).
  const float bc0 = bias[n0 + lm];
  const float bc1 = bias[n0 + 16 + lm];
  #pragma unroll
  for (int i = 0; i < 2; ++i) {
    const int rowb = m0 + i * 16 + lh * 8;
    const v8f& cA = i ? acc10 : acc00;    // j = 0 tile
    const v8f& cB = i ? acc11 : acc01;    // j = 1 tile
    #pragma unroll
    for (int r = 0; r < 8; ++r) {
      float v0 = cA[r] + bc0;
      float v1 = cB[r] + bc1;
      const size_t ro = (size_t)(rowb + r);
      if constexpr (MODE == 0) {
        outF[ro * ldo + n0 + lm]      = v0;
        outF[ro * ldo + n0 + 16 + lm] = v1;
      } else if constexpr (MODE == 1) {
        outB[ro * ldob + n0 + lm]      = f2bf(v0);
        outB[ro * ldob + n0 + 16 + lm] = f2bf(v1);
      } else {
        float t0 = tanhf(v0), t1 = tanhf(v1);
        outF[ro * ldo + n0 + lm]       = t0;
        outF[ro * ldo + n0 + 16 + lm]  = t1;
        outB[ro * ldob + n0 + lm]      = f2bf(t0);
        outB[ro * ldob + n0 + 16 + lm] = f2bf(t1);
      }
    }
  }
}

// ---------------------------------------------------------------------------
// LSTM pointwise: gates (B x 4H, [i|f|g|o]) -> c,h fp32 state; h also stored
// as bf16 into two staging destinations (column offset folded into pointers).
// ---------------------------------------------------------------------------
__global__ void lstm_pointwise(const float* __restrict__ gates,
                               float* __restrict__ c, float* __restrict__ h,
                               unsigned short* __restrict__ bdst1, int s1,
                               unsigned short* __restrict__ bdst2, int s2,
                               int Bn, int Hn)
{
  int idx = blockIdx.x * blockDim.x + threadIdx.x;
  if (idx >= Bn * Hn) return;
  int b = idx / Hn, j = idx - b * Hn;
  const float* g = gates + (size_t)b * 4 * Hn;
  float gi = sigmoidf_(g[j]);
  float gf = sigmoidf_(g[Hn + j]);
  float gg = tanhf(g[2 * Hn + j]);
  float go = sigmoidf_(g[3 * Hn + j]);
  float cn = gf * c[idx] + gi * gg;
  float hn = go * tanhf(cn);
  c[idx] = cn;
  h[idx] = hn;
  unsigned short hb = f2bf(hn);
  bdst1[(size_t)b * s1 + j] = hb;
  bdst2[(size_t)b * s2 + j] = hb;
}

// ---------------------------------------------------------------------------
// Attention: per batch element b (one 128-thread block):
//   e[l] = dot(h2[b], enc_proj[b,l]) (masked), alpha = softmax(e),
//   a_t[f] = sum_l alpha[l] * enc_hiddens[b,l,f]  -> bf16 into Xc[:, 0:1024]
// ---------------------------------------------------------------------------
__global__ __launch_bounds__(128)
void attn_kernel(const float* __restrict__ h2,
                 const unsigned short* __restrict__ encproj,  // (B*L, H) bf16
                 const float* __restrict__ ench,              // (B, L, HE) f32
                 const int* __restrict__ masks,               // (B, L)
                 unsigned short* __restrict__ aout, int aStride)
{
  const int b = blockIdx.x, tid = threadIdx.x;   // tid in [0,128)
  __shared__ float sh2[1024];
  __shared__ float e[128];
  __shared__ float red[128];

  const float* h2b = h2 + (size_t)b * 1024;
  #pragma unroll
  for (int i = 0; i < 8; ++i) sh2[tid + i * 128] = h2b[tid + i * 128];
  __syncthreads();

  // e_t
  const unsigned short* ep = encproj + ((size_t)b * 128 + tid) * 1024;
  float acc = 0.f;
  for (int k = 0; k < 1024; ++k) acc += sh2[k] * bf2f(ep[k]);
  if (masks[b * 128 + tid] != 0) acc = -__builtin_inff();
  e[tid] = acc;
  __syncthreads();

  // softmax over 128
  red[tid] = e[tid];
  __syncthreads();
  for (int s = 64; s > 0; s >>= 1) {
    if (tid < s) red[tid] = fmaxf(red[tid], red[tid + s]);
    __syncthreads();
  }
  float mx = red[0];
  __syncthreads();
  float ex = __expf(e[tid] - mx);
  red[tid] = ex;
  __syncthreads();
  for (int s = 64; s > 0; s >>= 1) {
    if (tid < s) red[tid] += red[tid + s];
    __syncthreads();
  }
  float inv = 1.0f / red[0];
  __syncthreads();
  e[tid] = ex * inv;                     // alpha
  __syncthreads();

  // a_t: 8 features per thread
  float a[8];
  #pragma unroll
  for (int j = 0; j < 8; ++j) a[j] = 0.f;
  const float* eb = ench + (size_t)b * 128 * 1024 + tid * 8;
  for (int l = 0; l < 128; ++l) {
    float al = e[l];
    const float* r = eb + (size_t)l * 1024;
    #pragma unroll
    for (int j = 0; j < 8; ++j) a[j] += al * r[j];
  }
  #pragma unroll
  for (int j = 0; j < 8; ++j)
    aout[(size_t)b * aStride + tid * 8 + j] = f2bf(a[j]);
}

// ---------------------------------------------------------------------------
// Small utility kernels
// ---------------------------------------------------------------------------
__global__ void copy2d_f32_bf16(const float* __restrict__ src, int sStride,
                                unsigned short* __restrict__ dst, int dStride,
                                int rows, int cols)
{
  long long total = (long long)rows * cols;
  long long stride = (long long)gridDim.x * blockDim.x;
  for (long long i = (long long)blockIdx.x * blockDim.x + threadIdx.x;
       i < total; i += stride) {
    long long r = i / cols, c = i - r * cols;
    dst[r * dStride + c] = f2bf(src[r * sStride + c]);
  }
}

__global__ void zero2d_bf16(unsigned short* __restrict__ dst, int dStride,
                            int rows, int cols)
{
  long long total = (long long)rows * cols;
  long long stride = (long long)gridDim.x * blockDim.x;
  for (long long i = (long long)blockIdx.x * blockDim.x + threadIdx.x;
       i < total; i += stride) {
    long long r = i / cols, c = i - r * cols;
    dst[r * dStride + c] = 0;
  }
}

__global__ void copy_f32(const float* __restrict__ s, float* __restrict__ d, int n)
{
  int i = blockIdx.x * blockDim.x + threadIdx.x;
  if (i < n) d[i] = s[i];
}

__global__ void bias_sum(const float* __restrict__ a, const float* __restrict__ b,
                         float* __restrict__ d, int n)
{
  int i = blockIdx.x * blockDim.x + threadIdx.x;
  if (i < n) d[i] = a[i] + b[i];
}

// ---------------------------------------------------------------------------
// Host driver
// ---------------------------------------------------------------------------
extern "C" void kernel_launch(void* const* d_in, const int* in_sizes, int n_in,
                              void* d_out, int out_size, void* d_ws, size_t ws_size,
                              hipStream_t stream)
{
  (void)in_sizes; (void)n_in; (void)out_size; (void)ws_size;
  const int B = 64, L = 128, T = 48, E = 512, HE = 1024, H = 1024, FH = 4096;
  const int K1 = E + H;        // 1536: [x | h1]
  const int K2 = 3 * H;        // 3072: [h1 | o_prev | h2]
  const int KC = HE + H;       // 2048: [a_t | h2]

  const float* enc_hiddens = (const float*)d_in[0];
  const int*   enc_masks   = (const int*)d_in[1];
  const float* h1_init = (const float*)d_in[2];
  const float* c1_init = (const float*)d_in[3];
  const float* h2_init = (const float*)d_in[4];
  const float* c2_init = (const float*)d_in[5];
  const float* captions = (const float*)d_in[6];
  const float* W_ih1 = (const float*)d_in[7];
  const float* W_hh1 = (const float*)d_in[8];
  const float* b_ih1 = (const float*)d_in[9];
  const float* b_hh1 = (const float*)d_in[10];
  const float* W_ih2 = (const float*)d_in[11];
  const float* W_hh2 = (const float*)d_in[12];
  const float* b_ih2 = (const float*)d_in[13];
  const float* b_hh2 = (const float*)d_in[14];
  const float* W_att  = (const float*)d_in[15];
  const float* b_att  = (const float*)d_in[16];
  const float* W_comb = (const float*)d_in[17];
  const float* b_comb = (const float*)d_in[18];
  float* out = (float*)d_out;

  // ---- workspace carve (256B aligned slabs) ----
  char* base = (char*)d_ws;
  size_t off = 0;
  auto take = [&](size_t bytes) -> char* {
    char* p = base + off;
    off += (bytes + 255) & ~(size_t)255;
    return p;
  };
  unsigned short* W1 = (unsigned short*)take((size_t)FH * K1 * 2); // [W_ih1|W_hh1]
  unsigned short* W2 = (unsigned short*)take((size_t)FH * K2 * 2); // [W_ih2|W_hh2]
  unsigned short* Wc = (unsigned short*)take((size_t)H * KC * 2);
  unsigned short* Wa = (unsigned short*)take((size_t)H * HE * 2);
  unsigned short* EH = (unsigned short*)take((size_t)B * L * HE * 2); // enc_hiddens bf16
  unsigned short* EP = (unsigned short*)take((size_t)B * L * H * 2);  // enc_proj bf16
  unsigned short* X1 = (unsigned short*)take((size_t)B * K1 * 2);
  unsigned short* X2 = (unsigned short*)take((size_t)B * K2 * 2);
  unsigned short* Xc = (unsigned short*)take((size_t)B * KC * 2);
  float* G1 = (float*)take((size_t)B * FH * 4);
  float* G2 = (float*)take((size_t)B * FH * 4);
  float* c1s = (float*)take((size_t)B * H * 4);
  float* c2s = (float*)take((size_t)B * H * 4);
  float* h1s = (float*)take((size_t)B * H * 4);
  float* h2s = (float*)take((size_t)B * H * 4);
  float* bias1 = (float*)take((size_t)FH * 4);
  float* bias2 = (float*)take((size_t)FH * 4);

  auto g2d = [](long long total) -> int {
    long long b = (total + 255) / 256;
    return (int)(b > 2048 ? 2048 : b);
  };

  // ---- one-time (per call) conversions ----
  copy2d_f32_bf16<<<g2d((long long)FH * E), 256, 0, stream>>>(W_ih1, E, W1, K1, FH, E);
  copy2d_f32_bf16<<<g2d((long long)FH * H), 256, 0, stream>>>(W_hh1, H, W1 + E, K1, FH, H);
  copy2d_f32_bf16<<<g2d((long long)FH * 2 * H), 256, 0, stream>>>(W_ih2, 2 * H, W2, K2, FH, 2 * H);
  copy2d_f32_bf16<<<g2d((long long)FH * H), 256, 0, stream>>>(W_hh2, H, W2 + 2 * H, K2, FH, H);
  copy2d_f32_bf16<<<g2d((long long)H * KC), 256, 0, stream>>>(W_comb, KC, Wc, KC, H, KC);
  copy2d_f32_bf16<<<g2d((long long)H * HE), 256, 0, stream>>>(W_att, HE, Wa, HE, H, HE);
  copy2d_f32_bf16<<<g2d((long long)B * L * HE), 256, 0, stream>>>(enc_hiddens, HE, EH, HE, B * L, HE);

  bias_sum<<<(FH + 255) / 256, 256, 0, stream>>>(b_ih1, b_hh1, bias1, FH);
  bias_sum<<<(FH + 255) / 256, 256, 0, stream>>>(b_ih2, b_hh2, bias2, FH);

  // ---- state init ----
  copy_f32<<<(B * H + 255) / 256, 256, 0, stream>>>(c1_init, c1s, B * H);
  copy_f32<<<(B * H + 255) / 256, 256, 0, stream>>>(c2_init, c2s, B * H);
  copy_f32<<<(B * H + 255) / 256, 256, 0, stream>>>(h2_init, h2s, B * H);
  copy2d_f32_bf16<<<g2d((long long)B * H), 256, 0, stream>>>(h1_init, H, X1 + E, K1, B, H);
  copy2d_f32_bf16<<<g2d((long long)B * H), 256, 0, stream>>>(h2_init, H, X2 + 2 * H, K2, B, H);
  zero2d_bf16<<<g2d((long long)B * H), 256, 0, stream>>>(X2 + H, K2, B, H);  // o0 = 0

  // ---- enc_proj = enc_hiddens @ W_att^T + b_att  (8192x1024, K=1024) ----
  {
    int waves = ((B * L) >> 5) * (H >> 5);   // 256 * 32 = 8192
    wmma_gemm<1><<<waves / 8, 256, 0, stream>>>(EH, HE, Wa, HE, b_att,
                                                nullptr, 0, EP, H, B * L, H, HE);
  }

  const int gateWaves = (B >> 5) * (FH >> 5);   // 2 * 128 = 256
  const int combWaves = (B >> 5) * (H >> 5);    // 2 * 32  = 64

  // ---- decode loop ----
  for (int t = 0; t < T; ++t) {
    // stage caption_t into X1[:, 0:E]
    copy2d_f32_bf16<<<g2d((long long)B * E), 256, 0, stream>>>(
        captions + (size_t)t * B * E, E, X1, K1, B, E);

    // cell 1 gates: (64 x 1536) x (1536 x 4096)
    wmma_gemm<0><<<gateWaves / 8, 256, 0, stream>>>(
        X1, K1, W1, K1, bias1, G1, FH, nullptr, 0, B, FH, K1);
    lstm_pointwise<<<(B * H + 255) / 256, 256, 0, stream>>>(
        G1, c1s, h1s, X1 + E, K1, X2, K2, B, H);

    // cell 2 gates: (64 x 3072) x (3072 x 4096)
    wmma_gemm<0><<<gateWaves / 8, 256, 0, stream>>>(
        X2, K2, W2, K2, bias2, G2, FH, nullptr, 0, B, FH, K2);
    lstm_pointwise<<<(B * H + 255) / 256, 256, 0, stream>>>(
        G2, c2s, h2s, X2 + 2 * H, K2, Xc + HE, KC, B, H);

    // attention -> a_t bf16 into Xc[:, 0:HE]
    attn_kernel<<<B, 128, 0, stream>>>(h2s, EP, enc_hiddens, enc_masks, Xc, KC);

    // o_t = tanh([a_t|h2] @ W_comb^T + b_comb); fp32 -> out[t], bf16 -> X2 o-slot
    wmma_gemm<2><<<combWaves / 8, 256, 0, stream>>>(
        Xc, KC, Wc, KC, b_comb, out + (size_t)t * B * H, H, X2 + H, K2, B, H, KC);
  }
}
